// MultiHeadAttention_83227876261908
// MI455X (gfx1250) — matrix operand order repaired and checked by reading, not compiled, not run on hardware
//
#include <hip/hip_runtime.h>
#include <stdint.h>

// ---------------------------------------------------------------------------
// MI455X (gfx1250) multi-head attention, bf16 WMMA pipeline.
//   S=4096, DIM=1024, H=16, DK=DV=64
// Stages: f32->bf16 convert | QKV GEMM (wmma bf16, double-buffered, async LDS)
//         | flash attention (wmma, TDM K-tile staging)
//         | output projection GEMM (wmma bf16, f32 out)
// ---------------------------------------------------------------------------

typedef __attribute__((ext_vector_type(16))) __bf16 v16bf;
typedef __attribute__((ext_vector_type(8)))  __bf16 v8bf;
typedef __attribute__((ext_vector_type(8)))  float  v8f;

typedef __attribute__((ext_vector_type(4))) unsigned int u32x4;
typedef __attribute__((ext_vector_type(8))) int          i32x8;
typedef __attribute__((ext_vector_type(4))) int          i32x4;

// Exact parameter type reported by the async-LDS builtin diagnostic.
typedef int vsi4 __attribute__((__vector_size__(16)));

#define AS3 __attribute__((address_space(3)))

#if __has_builtin(__builtin_amdgcn_global_load_async_to_lds_b128)
#define HAVE_ASYNC_LDS 1
#else
#define HAVE_ASYNC_LDS 0
#endif

#if __has_builtin(__builtin_amdgcn_tensor_load_to_lds)
#define HAVE_TDM 1
#else
#define HAVE_TDM 0
#endif

__device__ __forceinline__ unsigned short f2bf(float f) {
  unsigned u = __float_as_uint(f);
  u += 0x7fffu + ((u >> 16) & 1u);        // round-to-nearest-even
  return (unsigned short)(u >> 16);
}

// 16B global -> LDS copy; async (ASYNCcnt) when available.
__device__ __forceinline__ void copy_b128(const unsigned short* g, unsigned short* l) {
#if HAVE_ASYNC_LDS
  __builtin_amdgcn_global_load_async_to_lds_b128(
      (vsi4*)(uintptr_t)g, (vsi4*)(uintptr_t)l, 0, 0);
#else
  *reinterpret_cast<uint4*>(l) = *reinterpret_cast<const uint4*>(g);
#endif
}

__device__ __forceinline__ void async_wait() {
#if HAVE_ASYNC_LDS
#if __has_builtin(__builtin_amdgcn_s_wait_asynccnt)
  __builtin_amdgcn_s_wait_asynccnt(0);
#else
  asm volatile("s_wait_asynccnt 0x0" ::: "memory");
#endif
#endif
}

// Assemble a 16-element bf16 fragment from two 16-byte chunks (LDS or global).
__device__ __forceinline__ v16bf frag16(const unsigned short* base, int off0, int off1) {
  union { v16bf v; v8bf h[2]; } u;
  u.h[0] = *reinterpret_cast<const v8bf*>(base + off0);
  u.h[1] = *reinterpret_cast<const v8bf*>(base + off1);
  return u.v;
}

__device__ __forceinline__ v8f wmma_bf16(v16bf a, v16bf b, v8f c) {
  return __builtin_amdgcn_wmma_f32_16x16x32_bf16(false, a, false, b, (short)0, c,
                                                 false, false);
}

// ---------------------------------------------------------------------------
// Stage 1: f32 -> bf16 conversion (n multiple of 4)
// ---------------------------------------------------------------------------
__global__ __launch_bounds__(256) void cvt_bf16_kernel(const float* __restrict__ in,
                                                       unsigned short* __restrict__ out,
                                                       int n) {
  int i = (blockIdx.x * 256 + threadIdx.x) * 4;
  if (i + 3 < n) {
    float4 f = *reinterpret_cast<const float4*>(in + i);
    out[i + 0] = f2bf(f.x);
    out[i + 1] = f2bf(f.y);
    out[i + 2] = f2bf(f.z);
    out[i + 3] = f2bf(f.w);
  }
}

// ---------------------------------------------------------------------------
// Stages 2 & 4: bf16 GEMM  C[M,N] = A[M,K] * W[K,N] + bias
//   block tile 128x64, 8 waves of 32x32 (2x2 wmma tiles), K-step 32,
//   double-buffered LDS with async global->LDS staging for the A tile.
//   mode 0: out = bf16, scatter to [head][s][64], val=(acc+bias)*scale
//   mode 1: out = f32 row-major [M][N], val=acc+bias
// ---------------------------------------------------------------------------
#define GBM 128
#define GBN 64
#define GBK 32
#define GLDA 40   // (GBK + 8) elems -> conflict-free ds_load_b128
#define GLDB 40

__global__ __launch_bounds__(256) void gemm_bf16_kernel(
    const unsigned short* __restrict__ A,   // [M][Kdim] bf16
    const unsigned short* __restrict__ W,   // [Kdim][N] bf16
    const float* __restrict__ bias,         // [N]
    void* __restrict__ outp,
    int M, int Kdim, int N, int Sdim,
    int mode, float scale) {
  __shared__ unsigned short As[2][GBM * GLDA];
  __shared__ unsigned short Bs[2][GBN * GLDB];

  const int tid  = threadIdx.x;
  const int m0   = blockIdx.y * GBM;
  const int n0   = blockIdx.x * GBN;
  const int w    = tid >> 5;
  const int lane = tid & 31;
  const int lh   = lane & 15;
  const int half = lane >> 4;
  const int wm   = (w & 3) * 32;   // wave M offset within block tile
  const int wn   = (w >> 2) * 32;  // wave N offset

  v8f acc[2][2];
#pragma unroll
  for (int mt = 0; mt < 2; ++mt)
#pragma unroll
    for (int nt = 0; nt < 2; ++nt)
#pragma unroll
      for (int r = 0; r < 8; ++r) acc[mt][nt][r] = 0.0f;

  const int arow = tid >> 1;            // 0..127
  const int aseg = (tid & 1) * 16;      // elem offset (16 elems = 32B)

  auto stage = [&](int buf, int k0) {
    // ---- A tile (row-major [m][k]) : 128 x 32 bf16, async -> LDS ----
    const unsigned short* gA = A + (size_t)(m0 + arow) * Kdim + k0 + aseg;
    unsigned short* sA = &As[buf][arow * GLDA + aseg];
    copy_b128(gA, sA);
    copy_b128(gA + 8, sA + 8);
    if (k0 + GBK < Kdim) __builtin_prefetch(gA + GBK, 0, 0);
    // ---- B tile transposed into [n][k] : 64 x 32 bf16 ----
#pragma unroll
    for (int it = 0; it < 2; ++it) {
      int idx = tid + it * 256;          // 0..511
      int kk  = idx >> 4;                // 0..31
      int n4  = (idx & 15) * 4;          // 0..60
      const unsigned short* gW = W + (size_t)(k0 + kk) * N + n0 + n4;
      uint2 wv = *reinterpret_cast<const uint2*>(gW);
      if (k0 + GBK < Kdim) __builtin_prefetch(gW + (size_t)GBK * N, 0, 0);
      Bs[buf][(n4 + 0) * GLDB + kk] = (unsigned short)(wv.x & 0xffffu);
      Bs[buf][(n4 + 1) * GLDB + kk] = (unsigned short)(wv.x >> 16);
      Bs[buf][(n4 + 2) * GLDB + kk] = (unsigned short)(wv.y & 0xffffu);
      Bs[buf][(n4 + 3) * GLDB + kk] = (unsigned short)(wv.y >> 16);
    }
  };

  // prologue
  stage(0, 0);
  async_wait();
  __syncthreads();

  int buf = 0;
  for (int k0 = 0; k0 < Kdim; k0 += GBK, buf ^= 1) {
    if (k0 + GBK < Kdim) stage(buf ^ 1, k0 + GBK);   // overlap next tile

#pragma unroll
    for (int mt = 0; mt < 2; ++mt) {
      const unsigned short* ab = &As[buf][(wm + mt * 16 + lh) * GLDA];
      v16bf a = frag16(ab, half * 16, half * 16 + 8);
#pragma unroll
      for (int nt = 0; nt < 2; ++nt) {
        const unsigned short* bb = &Bs[buf][(wn + nt * 16 + lh) * GLDB];
        v16bf b = frag16(bb, half * 16, half * 16 + 8);
        acc[mt][nt] = wmma_bf16(a, b, acc[mt][nt]);
      }
    }
    async_wait();          // our async stores into buf^1 must land
    __syncthreads();       // everyone done reading buf / writing buf^1
  }

  // ---- epilogue ----
#pragma unroll
  for (int mt = 0; mt < 2; ++mt)
#pragma unroll
    for (int nt = 0; nt < 2; ++nt)
#pragma unroll
      for (int r = 0; r < 8; ++r) {
        int m = m0 + wm + mt * 16 + r + half * 8;
        int n = n0 + wn + nt * 16 + lh;
        float v = acc[mt][nt][r] + bias[n];
        if (mode == 0) {
          int head = n >> 6, dv = n & 63;
          ((unsigned short*)outp)[((size_t)(head * Sdim + m) << 6) + dv] =
              f2bf(v * scale);
        } else {
          ((float*)outp)[(size_t)m * N + n] = v;
        }
      }
}

// ---------------------------------------------------------------------------
// Stage 3: flash attention (bf16 wmma, f32 online softmax)
//   grid = (S/128, H); block 256 = 8 waves; each wave owns 16 query rows.
//   K tile staged by the Tensor Data Mover (TENSORcnt) when available,
//   overlapped with the manual transposed V staging.
// ---------------------------------------------------------------------------
#define FKT 32
#define KLD 72   // (64 + 8): TDM pad_interval=32 dwords, pad_amount=4 dwords
#define VLD 40   // (32 + 8)
#define PLD 40

__global__ __launch_bounds__(256) void flash_attn_kernel(
    const unsigned short* __restrict__ Q,   // [H][S][64] bf16 (pre-scaled)
    const unsigned short* __restrict__ K,   // [H][S][64] bf16
    const unsigned short* __restrict__ V,   // [H][S][64] bf16
    unsigned short* __restrict__ O,         // [S][H*64] bf16
    int S) {
  __shared__ unsigned short Ks[FKT * KLD];       // [key][dk]
  __shared__ unsigned short Vt[64 * VLD];        // [dv][key]   (transposed)
  __shared__ unsigned short Ps[8 * 16 * PLD];    // per-wave P tile [row][key]

  const int tid  = threadIdx.x;
  const int head = blockIdx.y;
  const int q0   = blockIdx.x * 128;
  const int w    = tid >> 5;
  const int lane = tid & 31;
  const int lh   = lane & 15;
  const int half = lane >> 4;

  // Q fragments stay resident for the whole pass (16 rows x 64 dk per wave).
  const int qrow = q0 + w * 16 + lh;
  const unsigned short* qb = Q + ((size_t)(head * S + qrow) << 6);
  v16bf qa0 = frag16(qb, half * 16, half * 16 + 8);        // dk 0..31
  v16bf qa1 = frag16(qb, 32 + half * 16, 32 + half * 16 + 8);   // dk 32..63

  float mrow[8], lrow[8];
  v8f o[4];
#pragma unroll
  for (int r = 0; r < 8; ++r) { mrow[r] = -1e30f; lrow[r] = 0.0f; }
#pragma unroll
  for (int d = 0; d < 4; ++d)
#pragma unroll
    for (int r = 0; r < 8; ++r) o[d][r] = 0.0f;

  unsigned short* myPs = &Ps[w * 16 * PLD];

  for (int j0 = 0; j0 < S; j0 += FKT) {
    const unsigned short* kTile = K + ((size_t)(head * S + j0) << 6);

#if HAVE_TDM
    // ---- K tile via Tensor Data Mover: 2D tile 64(dk) x 32(keys), bf16,
    //      LDS padding 4 dwords per 32-dword row -> KLD=72 layout. ----
    if (tid < 32) {
      unsigned long long ga = (unsigned long long)(uintptr_t)kTile;
      u32x4 g0 = {0u, 0u, 0u, 0u};
      g0[0] = 1u;                                          // count=1
      g0[1] = (unsigned)(uintptr_t)(&Ks[0]);               // lds_addr
      g0[2] = (unsigned)(ga & 0xffffffffu);                // global_addr lo
      g0[3] = (unsigned)((ga >> 32) & 0x01ffffffu) | (2u << 30);  // hi | type=2
      i32x8 g1 = {0, 0, 0, 0, 0, 0, 0, 0};
      g1[0] = (int)((1u << 16)        // data_size = 2 bytes
                    | (1u << 20)       // pad_enable
                    | (4u << 22)       // pad_interval = 32 dwords
                    | (3u << 25));     // pad_amount  = 4 dwords
      g1[1] = (int)(64u << 16);                            // tensor_dim0 = 64
      g1[2] = (int)(((unsigned)S & 0xffffu) << 16);        // tensor_dim1 lo
      g1[3] = (int)((64u << 16) | ((unsigned)S >> 16));    // tile_dim0=64 | td1 hi
      g1[4] = 32;                                          // tile_dim1 = 32 keys
      g1[5] = 64;                                          // tensor_dim0_stride
      i32x4 z4 = {0, 0, 0, 0};
      i32x8 z8 = {0, 0, 0, 0, 0, 0, 0, 0};
      __builtin_amdgcn_tensor_load_to_lds(g0, g1, z4, z4, z8, 0);
    }
#else
    // ---- K tile fallback: 32 keys x 64 dk, 16B per thread ----
    {
      int jr  = tid >> 3;             // 0..31
      int seg = (tid & 7) * 8;        // elem offset, 16B chunks
      copy_b128(kTile + ((size_t)jr << 6) + seg, &Ks[jr * KLD + seg]);
    }
#endif

    // ---- stage V tile transposed: Vt[dv][key] (overlaps TDM) ----
#pragma unroll
    for (int it = 0; it < 2; ++it) {
      int idx = tid + it * 256;       // 0..511
      int key = idx >> 4;             // 0..31
      int d4  = (idx & 15) * 4;       // 0..60
      const unsigned short* gV = V + ((size_t)(head * S + j0 + key) << 6) + d4;
      uint2 dv = *reinterpret_cast<const uint2*>(gV);
      if (j0 + FKT < S) __builtin_prefetch(gV + (FKT << 6), 0, 0);
      Vt[(d4 + 0) * VLD + key] = (unsigned short)(dv.x & 0xffffu);
      Vt[(d4 + 1) * VLD + key] = (unsigned short)(dv.x >> 16);
      Vt[(d4 + 2) * VLD + key] = (unsigned short)(dv.y & 0xffffu);
      Vt[(d4 + 3) * VLD + key] = (unsigned short)(dv.y >> 16);
    }

#if HAVE_TDM
    if (tid < 32) __builtin_amdgcn_s_wait_tensorcnt(0);
#else
    async_wait();
#endif
    __syncthreads();

    // ---- S = Q @ K^T : 16 x 32 scores per wave ----
    v8f s0, s1;
#pragma unroll
    for (int r = 0; r < 8; ++r) { s0[r] = 0.0f; s1[r] = 0.0f; }
    {
      const unsigned short* kb0 = &Ks[lh * KLD];           // keys 0..15
      const unsigned short* kb1 = &Ks[(16 + lh) * KLD];    // keys 16..31
      s0 = wmma_bf16(qa0, frag16(kb0, half * 16, half * 16 + 8), s0);
      s0 = wmma_bf16(qa1, frag16(kb0, 32 + half * 16, 32 + half * 16 + 8), s0);
      s1 = wmma_bf16(qa0, frag16(kb1, half * 16, half * 16 + 8), s1);
      s1 = wmma_bf16(qa1, frag16(kb1, 32 + half * 16, 32 + half * 16 + 8), s1);
    }

    // ---- online softmax (rows live across 16 lanes of each half) ----
#pragma unroll
    for (int r = 0; r < 8; ++r) {
      float mx = fmaxf(s0[r], s1[r]);
#pragma unroll
      for (int msk = 1; msk <= 8; msk <<= 1)
        mx = fmaxf(mx, __shfl_xor(mx, msk, 32));
      float mnew  = fmaxf(mrow[r], mx);
      float alpha = __expf(mrow[r] - mnew);
      float p0 = __expf(s0[r] - mnew);
      float p1 = __expf(s1[r] - mnew);
      float rs = p0 + p1;
#pragma unroll
      for (int msk = 1; msk <= 8; msk <<= 1)
        rs += __shfl_xor(rs, msk, 32);
      lrow[r] = lrow[r] * alpha + rs;
      mrow[r] = mnew;
#pragma unroll
      for (int d = 0; d < 4; ++d) o[d][r] *= alpha;
      myPs[(r + half * 8) * PLD + lh]      = f2bf(p0);
      myPs[(r + half * 8) * PLD + 16 + lh] = f2bf(p1);
    }

    // ---- O += P @ V  (P as A-fragment from LDS, Vt as B-fragment) ----
    v16bf pa = frag16(&myPs[lh * PLD], half * 16, half * 16 + 8);
#pragma unroll
    for (int d = 0; d < 4; ++d) {
      const unsigned short* vb = &Vt[(d * 16 + lh) * VLD];
      o[d] = wmma_bf16(pa, frag16(vb, half * 16, half * 16 + 8), o[d]);
    }
    __syncthreads();
  }

  // ---- normalize and store: O[s][head*64 + dv] ----
#pragma unroll
  for (int d = 0; d < 4; ++d)
#pragma unroll
    for (int r = 0; r < 8; ++r) {
      int srow = q0 + w * 16 + r + half * 8;
      int col  = (head << 6) + d * 16 + lh;
      O[(size_t)srow * 1024 + col] = f2bf(o[d][r] / lrow[r]);
    }
}

// ---------------------------------------------------------------------------
// Launcher
// ---------------------------------------------------------------------------
extern "C" void kernel_launch(void* const* d_in, const int* in_sizes, int n_in,
                              void* d_out, int out_size, void* d_ws, size_t ws_size,
                              hipStream_t stream) {
  (void)in_sizes; (void)n_in; (void)out_size; (void)ws_size;
  const int S = 4096, D = 1024, H = 16;

  const float* x  = (const float*)d_in[0];
  const float* Wq = (const float*)d_in[1];
  const float* bq = (const float*)d_in[2];
  const float* Wk = (const float*)d_in[3];
  const float* bk = (const float*)d_in[4];
  const float* Wv = (const float*)d_in[5];
  const float* bv = (const float*)d_in[6];
  const float* Wo = (const float*)d_in[7];
  const float* bo = (const float*)d_in[8];
  float* out = (float*)d_out;

  char* ws = (char*)d_ws;
  const size_t MB = 1024u * 1024u;
  unsigned short* xb  = (unsigned short*)(ws + 0 * MB);    // 8 MB  [S][D]
  unsigned short* Wqb = (unsigned short*)(ws + 8 * MB);    // 2 MB
  unsigned short* Wkb = (unsigned short*)(ws + 10 * MB);
  unsigned short* Wvb = (unsigned short*)(ws + 12 * MB);
  unsigned short* Wob = (unsigned short*)(ws + 14 * MB);
  unsigned short* Qb  = (unsigned short*)(ws + 16 * MB);   // 8 MB [H][S][64]
  unsigned short* Kb  = (unsigned short*)(ws + 24 * MB);
  unsigned short* Vb  = (unsigned short*)(ws + 32 * MB);
  unsigned short* Ob  = (unsigned short*)(ws + 40 * MB);   // 8 MB [S][D]

  // Stage 1: convert to bf16
  cvt_bf16_kernel<<<(S * D) / 1024, 256, 0, stream>>>(x, xb, S * D);
  cvt_bf16_kernel<<<(D * D) / 1024, 256, 0, stream>>>(Wq, Wqb, D * D);
  cvt_bf16_kernel<<<(D * D) / 1024, 256, 0, stream>>>(Wk, Wkb, D * D);
  cvt_bf16_kernel<<<(D * D) / 1024, 256, 0, stream>>>(Wv, Wvb, D * D);
  cvt_bf16_kernel<<<(D * D) / 1024, 256, 0, stream>>>(Wo, Wob, D * D);

  // Stage 2: Q/K/V projections (scale 1/sqrt(DK) folded into Q)
  dim3 gg(D / GBN, S / GBM);
  gemm_bf16_kernel<<<gg, 256, 0, stream>>>(xb, Wqb, bq, Qb, S, D, D, S, 0, 0.125f);
  gemm_bf16_kernel<<<gg, 256, 0, stream>>>(xb, Wkb, bk, Kb, S, D, D, S, 0, 1.0f);
  gemm_bf16_kernel<<<gg, 256, 0, stream>>>(xb, Wvb, bv, Vb, S, D, D, S, 0, 1.0f);

  // Stage 3: flash attention
  flash_attn_kernel<<<dim3(S / 128, H), 256, 0, stream>>>(Qb, Kb, Vb, Ob, S);

  // Stage 4: output projection -> f32
  gemm_bf16_kernel<<<gg, 256, 0, stream>>>(Ob, Wob, bo, out, S, D, D, S, 1, 1.0f);
}
